// GGNN_76665166233620
// MI455X (gfx1250) — matrix operand order repaired and checked by reading, not compiled, not run on hardware
//
#include <hip/hip_runtime.h>
#include <hip/hip_bf16.h>
#include <math.h>
#include <stdint.h>

// GGNN for MI455X (gfx1250), wave32, one workgroup per batch element.
// H = hidden @ W_in[0:64,:] kept incrementally in LDS; rel contribution is a
// 51x64 LDS lookup table (rel_table @ W_in[64:96,:]).  The only real GEMM
// (H0 init) runs on v_wmma_f32_16x16x32_f16.  The per-step rel_matrix row is
// double-buffered into LDS with GLOBAL_LOAD_ASYNC_TO_LDS_B32 (ASYNCcnt), so
// its global-memory latency is hidden behind the previous step's compute.

typedef __attribute__((ext_vector_type(16))) _Float16 v16h;
typedef __attribute__((ext_vector_type(8)))  float    v8f;

#define Bx 128
#define Nx 128
#define Ex 64
#define Tx 32
#define Rx 32
#define GI 160      // T + E + E
#define NTHREADS 256

__device__ __forceinline__ float sigm(float x) { return 1.0f / (1.0f + __expf(-x)); }

__global__ __launch_bounds__(NTHREADS)
void ggnn_kernel(const int* __restrict__ node_slice, const int* __restrict__ type_slice,
                 const float* __restrict__ distance, const int* __restrict__ rel_matrix,
                 const int* __restrict__ input_length,
                 const float* __restrict__ emb_table, const float* __restrict__ type_table,
                 const float* __restrict__ rel_table,
                 const float* __restrict__ W_in, const float* __restrict__ b_in,
                 const float* __restrict__ Wr,   const float* __restrict__ br,
                 const float* __restrict__ Wz,   const float* __restrict__ bz,
                 const float* __restrict__ Wt,   const float* __restrict__ bt,
                 const float* __restrict__ W_co, const float* __restrict__ b_co,
                 const float* __restrict__ W1,   const float* __restrict__ b1,
                 const float* __restrict__ W2,   const float* __restrict__ b2,
                 const float* __restrict__ W3,   const float* __restrict__ b3,
                 const float* __restrict__ W4,   const float* __restrict__ b4,
                 float* __restrict__ out, float* __restrict__ wsH)
{
    __shared__ float sh_H[Nx * Ex];        // hidden @ W_in[0:64,:]   (32 KB, hot)
    __shared__ float sh_rp[52 * Ex];       // rel projection table    (13 KB, hot)
    __shared__ float sh_disc[Nx];
    __shared__ int   sh_relbuf[2][Nx];     // double-buffered rel row (async-to-LDS)
    __shared__ float sh_bin[Ex];
    __shared__ float sh_a[GI];
    __shared__ float sh_dis[Ex];
    __shared__ float sh_r[Ex];             // r gate, later reused as h_hat
    __shared__ float sh_z[Ex];
    __shared__ float sh_new[Ex];
    __shared__ float sh_red[NTHREADS];
    __shared__ float sh_red2[NTHREADS];
    // readout scratch
    __shared__ float sh_user[Ex], sh_item[Ex], sh_sub[Ex], sh_ua[Ex], sh_ia[Ex];
    __shared__ float sh_in[128], sh_temp[192], sh_h1[128], sh_h2[64], sh_h3[32];

    const int b    = blockIdx.x;
    const int t    = threadIdx.x;
    const int lane = t & 31;
    const int wid  = t >> 5;               // wave id 0..7
    const int c    = t & 63;               // channel
    const int g    = t >> 6;               // 0..3 k-group / n-group
    const int len_b = input_length[b];
    float* hid = wsH + (size_t)b * (Nx * Ex);

    // ---------------- init ----------------
    if (t < Nx) sh_disc[t] = -0.1f * distance[b * Nx + t];
    if (t < Ex) sh_bin[t]  = b_in[t];

    // prologue: async-stage relation row 0 into buffer 0 (waves 0-3)
    if (t < Nx) {
        uint32_t lds_a = (uint32_t)(uintptr_t)&sh_relbuf[0][t];
        const int* ga  = rel_matrix + (size_t)b * Nx * Nx + t;
        asm volatile("global_load_async_to_lds_b32 %0, %1, off"
                     :: "v"(lds_a), "v"(ga) : "memory");
    }

    // stage hidden0 = emb_table[node_slice] into global workspace (fp32 state)
    for (int idx = t; idx < Nx * Ex; idx += NTHREADS) {
        int n = idx >> 6, cc = idx & 63;
        hid[idx] = emb_table[(size_t)node_slice[b * Nx + n] * Ex + cc];
    }

    // rel projection table: rp[v][c] = sum_k rel_table[v][k] * W_in[64+k][c]
    for (int idx = t; idx < 51 * Ex; idx += NTHREADS) {
        int v = idx >> 6, cc = idx & 63;
        float s = 0.f;
        #pragma unroll
        for (int k = 0; k < Rx; ++k)
            s += rel_table[v * Rx + k] * W_in[(Ex + k) * Ex + cc];
        sh_rp[idx] = s;
    }

    // H0 = hidden0 @ W_in[0:64,:] via WMMA (f16 in, f32 accum).
    // wave `wid` owns M-tile wid; 4 N-tiles x 2 K-chunks of 32.
    {
        const int mrow = (wid << 4) + (lane & 15);
        const float* arow = emb_table + (size_t)node_slice[b * Nx + mrow] * Ex;
        v8f acc[4] = {};
        #pragma unroll
        for (int kc = 0; kc < 2; ++kc) {
            v16h A;
            #pragma unroll
            for (int s = 0; s < 16; ++s) {
                // ISA 16-bit A 16x32 layout: lanes 0-15 K=0-7/16-23, lanes 16-31 K=8-15/24-31
                int k = s + ((s >= 8) ? 8 : 0) + ((lane >= 16) ? 8 : 0) + kc * 32;
                A[s] = (_Float16)arow[k];
            }
            #pragma unroll
            for (int nt = 0; nt < 4; ++nt) {
                v16h Bv;
                #pragma unroll
                for (int s = 0; s < 16; ++s) {
                    // B 32x16 layout: lanes 0-15 K=0-7/16-23, lanes 16-31 K=8-15/24-31
                    int k = s + ((s >= 8) ? 8 : 0) + ((lane >= 16) ? 8 : 0) + kc * 32;
                    Bv[s] = (_Float16)W_in[k * Ex + nt * 16 + (lane & 15)];
                }
                acc[nt] = __builtin_amdgcn_wmma_f32_16x16x32_f16(
                    false, A, false, Bv, (short)0, acc[nt], false, false);
            }
        }
        #pragma unroll
        for (int nt = 0; nt < 4; ++nt)
            #pragma unroll
            for (int r = 0; r < 8; ++r) {
                int ml = r + ((lane >= 16) ? 8 : 0);   // C/D layout: VGPR r -> M=r / r+8
                sh_H[((wid << 4) + ml) * Ex + nt * 16 + (lane & 15)] = acc[nt][r];
            }
    }

    // ---------------- 128-step recurrence ----------------
    for (int i = 0; i < Nx; ++i) {
        const int cur = i & 1;
        // issue async stage of row i+1 into the other buffer, then wait for
        // row i only (async loads retire in order per wave -> asynccnt<=1).
        if (i + 1 < Nx) {
            if (t < Nx) {
                uint32_t lds_a = (uint32_t)(uintptr_t)&sh_relbuf[cur ^ 1][t];
                const int* ga  = rel_matrix + ((size_t)b * Nx + i + 1) * Nx + t;
                asm volatile("global_load_async_to_lds_b32 %0, %1, off"
                             :: "v"(lds_a), "v"(ga) : "memory");
            }
            asm volatile("s_wait_asynccnt 0x1" ::: "memory");
        } else {
            asm volatile("s_wait_asynccnt 0x0" ::: "memory");
        }
        __syncthreads();

        // dis_pre[c] = max over valid n of relu(H[n][c]+rp[rel][c]+b_in[c]) * disc[n]
        float m = -INFINITY, anyv = 0.f;
        for (int n = g; n < Nx; n += 4) {
            int rv = sh_relbuf[cur][n];
            if (rv != 0 && n < len_b) {
                float pre = sh_H[n * Ex + c] + sh_rp[rv * Ex + c] + sh_bin[c];
                pre = fmaxf(pre, 0.f);
                m = fmaxf(m, pre * sh_disc[n]);
                anyv = 1.f;
            }
        }
        sh_red[t] = m; sh_red2[t] = anyv;
        __syncthreads();
        if (t < Ex) {
            float mm = fmaxf(fmaxf(sh_red[t],      sh_red[t + 64]),
                             fmaxf(sh_red[t + 128], sh_red[t + 192]));
            float av = fmaxf(fmaxf(sh_red2[t],      sh_red2[t + 64]),
                             fmaxf(sh_red2[t + 128], sh_red2[t + 192]));
            sh_dis[t] = (av > 0.f) ? mm : 0.f;
        } else if (t < 96) {             // a[0:32] = type_emb[b][i]
            int ti = type_slice[b * Nx + i];
            sh_a[t - 64] = type_table[ti * Tx + (t - 64)];
        } else if (t < 160) {            // a[32:96] = node_emb[b][i]
            int ni = node_slice[b * Nx + i];
            sh_a[t - 64] = emb_table[(size_t)ni * Ex + (t - 96)];
        }
        __syncthreads();
        if (t < Ex) sh_a[96 + t] = sh_dis[t];
        __syncthreads();

        // r gate
        {
            float p = 0.f; const int k0 = g * 40;
            for (int k = 0; k < 40; ++k) p += sh_a[k0 + k] * Wr[(k0 + k) * Ex + c];
            sh_red[t] = p; __syncthreads();
            if (t < Ex)
                sh_r[t] = sigm(sh_red[t] + sh_red[t+64] + sh_red[t+128] + sh_red[t+192] + br[t]);
            __syncthreads();
        }
        // z gate (reads sh_a with dis tail, so before the tail rewrite)
        {
            float p = 0.f; const int k0 = g * 40;
            for (int k = 0; k < 40; ++k) p += sh_a[k0 + k] * Wz[(k0 + k) * Ex + c];
            sh_red[t] = p; __syncthreads();
            if (t < Ex)
                sh_z[t] = sigm(sh_red[t] + sh_red[t+64] + sh_red[t+128] + sh_red[t+192] + bz[t]);
            __syncthreads();
        }
        // a2 tail: r * dis_pre
        if (t < Ex) sh_a[96 + t] = sh_r[t] * sh_dis[t];
        __syncthreads();
        // h_hat (reuse sh_r)
        {
            float p = 0.f; const int k0 = g * 40;
            for (int k = 0; k < 40; ++k) p += sh_a[k0 + k] * Wt[(k0 + k) * Ex + c];
            sh_red[t] = p; __syncthreads();
            if (t < Ex)
                sh_r[t] = tanhf(sh_red[t] + sh_red[t+64] + sh_red[t+128] + sh_red[t+192] + bt[t]);
            __syncthreads();
        }

        // hidden row update
        const bool active = (i < len_b);   // uniform across block
        if (t < Ex) {
            float dv = sh_dis[t], zz = sh_z[t];
            float up = (1.f - zz) * dv + zz * sh_r[t];
            float nv = active ? up : hid[i * Ex + t];
            hid[i * Ex + t] = nv;
            sh_new[t] = nv;
        }
        __syncthreads();
        if (active) {                      // re-project updated row into H
            float p = 0.f; const int k0 = g * 16;
            #pragma unroll
            for (int k = 0; k < 16; ++k) p += sh_new[k0 + k] * W_in[(k0 + k) * Ex + c];
            sh_red[t] = p; __syncthreads();
            if (t < Ex)
                sh_H[i * Ex + t] = sh_red[t] + sh_red[t+64] + sh_red[t+128] + sh_red[t+192];
            __syncthreads();
        }
    }
    __syncthreads();

    // ---------------- readout ----------------
    if (t < Ex) { sh_user[t] = hid[t]; sh_item[t] = hid[(len_b - 1) * Ex + t]; }
    {
        float m = -INFINITY;
        for (int n = g; n < len_b; n += 4) m = fmaxf(m, hid[n * Ex + c]);
        sh_red[t] = m; __syncthreads();
        if (t < Ex)
            sh_sub[t] = fmaxf(fmaxf(sh_red[t], sh_red[t+64]), fmaxf(sh_red[t+128], sh_red[t+192]));
        __syncthreads();
    }
    // ua = relu([user||sub] @ W_co + b_co)
    if (t < 64) sh_in[t] = sh_user[t]; else if (t < 128) sh_in[t] = sh_sub[t - 64];
    __syncthreads();
    {
        float p = 0.f; const int k0 = g * 32;
        for (int k = 0; k < 32; ++k) p += sh_in[k0 + k] * W_co[(k0 + k) * Ex + c];
        sh_red[t] = p; __syncthreads();
        if (t < Ex)
            sh_ua[t] = fmaxf(sh_red[t] + sh_red[t+64] + sh_red[t+128] + sh_red[t+192] + b_co[t], 0.f);
        __syncthreads();
    }
    // ia = relu([item||sub] @ W_co + b_co)
    if (t < 64) sh_in[t] = sh_item[t];
    __syncthreads();
    {
        float p = 0.f; const int k0 = g * 32;
        for (int k = 0; k < 32; ++k) p += sh_in[k0 + k] * W_co[(k0 + k) * Ex + c];
        sh_red[t] = p; __syncthreads();
        if (t < Ex)
            sh_ia[t] = fmaxf(sh_red[t] + sh_red[t+64] + sh_red[t+128] + sh_red[t+192] + b_co[t], 0.f);
        __syncthreads();
    }
    if (t < 64)       sh_temp[t] = sh_ua[t] * sh_user[t];
    else if (t < 128) sh_temp[t] = sh_sub[t - 64];
    else if (t < 192) sh_temp[t] = sh_ia[t - 128] * sh_item[t - 128];
    __syncthreads();
    // h1 = relu(temp(192) @ W1(192,128) + b1)
    {
        int j = t & 127, g2 = t >> 7;
        float p = 0.f; const int k0 = g2 * 96;
        for (int k = 0; k < 96; ++k) p += sh_temp[k0 + k] * W1[(k0 + k) * 128 + j];
        sh_red[t] = p; __syncthreads();
        if (t < 128) sh_h1[t] = fmaxf(sh_red[t] + sh_red[t + 128] + b1[t], 0.f);
        __syncthreads();
    }
    // h2 = relu(h1 @ W2(128,64) + b2)
    {
        float p = 0.f; const int k0 = g * 32;
        for (int k = 0; k < 32; ++k) p += sh_h1[k0 + k] * W2[(k0 + k) * 64 + c];
        sh_red[t] = p; __syncthreads();
        if (t < 64)
            sh_h2[t] = fmaxf(sh_red[t] + sh_red[t+64] + sh_red[t+128] + sh_red[t+192] + b2[t], 0.f);
        __syncthreads();
    }
    // h3 = relu(h2 @ W3(64,32) + b3)
    {
        int j = t & 31, g3 = t >> 5;
        float p = 0.f; const int k0 = g3 * 8;
        for (int k = 0; k < 8; ++k) p += sh_h2[k0 + k] * W3[(k0 + k) * 32 + j];
        sh_red[t] = p; __syncthreads();
        if (t < 32) {
            float s = 0.f;
            #pragma unroll
            for (int q = 0; q < 8; ++q) s += sh_red[t + q * 32];
            sh_h3[t] = fmaxf(s + b3[t], 0.f);
        }
        __syncthreads();
    }
    if (t == 0) {
        float s = b4[0];
        #pragma unroll
        for (int k = 0; k < 32; ++k) s += sh_h3[k] * W4[k];
        out[b] = sigm(s);
    }
}

extern "C" void kernel_launch(void* const* d_in, const int* in_sizes, int n_in,
                              void* d_out, int out_size, void* d_ws, size_t ws_size,
                              hipStream_t stream) {
    (void)in_sizes; (void)n_in; (void)out_size; (void)ws_size;
    const int*   node_slice = (const int*)d_in[0];
    const int*   type_slice = (const int*)d_in[1];
    const float* distance   = (const float*)d_in[2];
    const int*   rel_matrix = (const int*)d_in[3];
    const int*   input_len  = (const int*)d_in[4];
    // d_in[5] = isBatch (unused)
    const float* emb_table  = (const float*)d_in[6];
    const float* type_table = (const float*)d_in[7];
    const float* rel_table  = (const float*)d_in[8];
    const float* W_in = (const float*)d_in[9];  const float* b_in = (const float*)d_in[10];
    const float* Wr   = (const float*)d_in[11]; const float* br   = (const float*)d_in[12];
    const float* Wz   = (const float*)d_in[13]; const float* bz   = (const float*)d_in[14];
    const float* Wt   = (const float*)d_in[15]; const float* bt   = (const float*)d_in[16];
    const float* W_co = (const float*)d_in[17]; const float* b_co = (const float*)d_in[18];
    const float* W1   = (const float*)d_in[19]; const float* b1   = (const float*)d_in[20];
    const float* W2   = (const float*)d_in[21]; const float* b2   = (const float*)d_in[22];
    const float* W3   = (const float*)d_in[23]; const float* b3   = (const float*)d_in[24];
    const float* W4   = (const float*)d_in[25]; const float* b4   = (const float*)d_in[26];

    ggnn_kernel<<<dim3(Bx), dim3(NTHREADS), 0, stream>>>(
        node_slice, type_slice, distance, rel_matrix, input_len,
        emb_table, type_table, rel_table,
        W_in, b_in, Wr, br, Wz, bz, Wt, bt, W_co, b_co,
        W1, b1, W2, b2, W3, b3, W4, b4,
        (float*)d_out, (float*)d_ws);
}
